// DistGAT_45621142618443
// MI455X (gfx1250) — matrix operand order repaired and checked by reading, not compiled, not run on hardware
//
#include <hip/hip_runtime.h>
#include <math.h>

#define HEADS 4
#define FDIM 64
#define OUTF 256   // HEADS*FDIM
#define INFEATS 256
#define NEG_SLOPE 0.2f

typedef __attribute__((ext_vector_type(16))) __bf16          v16bf;
typedef __attribute__((ext_vector_type(8)))  float           v8f;

// ---------------------------------------------------------------- helpers ---
__device__ __forceinline__ unsigned int f2bf_pk(float lo, float hi) {
  // pack two f32 -> two bf16 (RNE) in one dword
  union { float f; unsigned u; } a, b; a.f = lo; b.f = hi;
  unsigned ra = a.u + 0x7fffu + ((a.u >> 16) & 1u);
  unsigned rb = b.u + 0x7fffu + ((b.u >> 16) & 1u);
  return (ra >> 16) | (rb & 0xffff0000u);
}

union Frag {
  v16bf v;
  uint4 q[2];   // q[0] = elements 0..7, q[1] = elements 8..15
};

// Load a 16-element bf16 fragment from two contiguous 8-element runs.
__device__ __forceinline__ v16bf load_frag2(const unsigned short* p0,
                                            const unsigned short* p1) {
  Frag f;
  f.q[0] = *(const uint4*)p0;
  f.q[1] = *(const uint4*)p1;
  return f.v;
}

// ------------------------------------------------------------- kernel A -----
// One-time f32 -> bf16 conversion (8 elements / thread, packed b128 store)
__global__ void k_cvt_bf16(const float* __restrict__ in,
                           unsigned short* __restrict__ outp, int nElem) {
  int base = (blockIdx.x * blockDim.x + threadIdx.x) * 8;
  if (base >= nElem) return;
  const float4* p = (const float4*)(in + base);
  float4 a = p[0], b = p[1];
  uint4 r;
  r.x = f2bf_pk(a.x, a.y);
  r.y = f2bf_pk(a.z, a.w);
  r.z = f2bf_pk(b.x, b.y);
  r.w = f2bf_pk(b.z, b.w);
  *(uint4*)(outp + base) = r;
}

// ------------------------------------------------------------- kernel 0 -----
// out = bias broadcast, denom = 0
__global__ void k_init(float* __restrict__ out, float* __restrict__ denom,
                       const float* __restrict__ bias, int n) {
  int i = blockIdx.x * blockDim.x + threadIdx.x;
  if (i < n * OUTF) out[i] = bias[i & (OUTF - 1)];
  if (i < n * HEADS) denom[i] = 0.0f;
}

// ------------------------------------------------------------- kernel 1 -----
// h[n,256] = feat[n,256] @ W[256,256]^T   via v_wmma_f32_16x16x32_bf16
// Block = one 64-col strip x 8 row-tiles (one per wave).
// W strip (64x256 bf16 = 32KB) staged in LDS once per block; B fragments are
// then ds_load_b128 pairs, A fragments global b128 pairs.
// A frag (16x32, lane = half*16+m): elem j -> K = kb + (j<8?0:16) + half*8 + j%8
// B frag (32x16, lane n = lane%16): elem j -> K = kb + (j<8?0:16) + half*8 + j%8
__global__ void __launch_bounds__(256)
k_gemm_wmma(const unsigned short* __restrict__ featb,
            const unsigned short* __restrict__ Wb,
            float* __restrict__ hmat, int n) {
  __shared__ unsigned short Wl[64 * INFEATS];   // 32 KB

  const int lane = threadIdx.x & 31;
  const int half = lane >> 4;
  const int sub  = lane & 15;
  const int wv   = threadIdx.x >> 5;            // wave in block: 0..7
  const int tiles_m = n >> 4;                   // N multiple of 16
  const int strip = blockIdx.x & 3;             // 4 column strips of 64
  const int mtg   = blockIdx.x >> 2;            // group of 8 row-tiles
  const int col0  = strip << 6;

  // cooperative copy: 64 rows x 256 k of bf16 = 2048 uint4
  {
    const uint4* gsrc = (const uint4*)(Wb + (size_t)col0 * INFEATS);
    uint4* ldst = (uint4*)Wl;
    for (int i = threadIdx.x; i < 2048; i += 256) ldst[i] = gsrc[i];
  }
  __syncthreads();

  const int mt = (mtg << 3) + wv;
  if (mt >= tiles_m) return;
  const int row0 = mt << 4;

  const unsigned short* arow = featb + (size_t)(row0 + sub) * INFEATS;

  const v8f vzero = {0.f, 0.f, 0.f, 0.f, 0.f, 0.f, 0.f, 0.f};
  v8f acc[4];
  acc[0] = vzero; acc[1] = vzero; acc[2] = vzero; acc[3] = vzero;

#pragma unroll
  for (int kb = 0; kb < INFEATS; kb += 32) {
    v16bf afrag = load_frag2(arow + kb + half * 8,
                             arow + kb + 16 + half * 8);
#pragma unroll
    for (int t = 0; t < 4; ++t) {
      const unsigned short* wrow =
          Wl + (size_t)(t * 16 + sub) * INFEATS + kb;
      v16bf bfrag = load_frag2(wrow + half * 8, wrow + 16 + half * 8);
      // (neg_a, A, neg_b, B, c_mod, C, reuse_a, reuse_b)
      acc[t] = __builtin_amdgcn_wmma_f32_16x16x32_bf16(
          false, afrag, false, bfrag, (short)0, acc[t], false, false);
    }
  }

  // C/D layout: VGPR r -> row r + half*8, col = sub (within 16x16 tile)
#pragma unroll
  for (int t = 0; t < 4; ++t) {
    float* obase = hmat + (size_t)(row0 + half * 8) * OUTF + col0 + t * 16 + sub;
#pragma unroll
    for (int r = 0; r < 8; ++r) obase[(size_t)r * OUTF] = acc[t][r];
  }
}

// ------------------------------------------------------------- kernel 2 -----
// el[n,h] = sum_f h[n,h,f]*attn_l[h,f]  (and er).  One wave per node.
__global__ void k_scores(const float* __restrict__ hmat,
                         const float* __restrict__ attn_l,
                         const float* __restrict__ attn_r,
                         float* __restrict__ el, float* __restrict__ er, int n) {
  const int wave = (blockIdx.x * blockDim.x + threadIdx.x) >> 5;
  const int lane = threadIdx.x & 31;
  if (wave >= n) return;
  const float* row = hmat + (size_t)wave * OUTF;
  float pl[HEADS], pr[HEADS];
#pragma unroll
  for (int hd = 0; hd < HEADS; ++hd) {
    float x0 = row[hd * FDIM + lane];
    float x1 = row[hd * FDIM + 32 + lane];
    pl[hd] = x0 * attn_l[hd * FDIM + lane] + x1 * attn_l[hd * FDIM + 32 + lane];
    pr[hd] = x0 * attn_r[hd * FDIM + lane] + x1 * attn_r[hd * FDIM + 32 + lane];
  }
#pragma unroll
  for (int off = 16; off > 0; off >>= 1) {
#pragma unroll
    for (int hd = 0; hd < HEADS; ++hd) {
      pl[hd] += __shfl_down(pl[hd], off, 32);
      pr[hd] += __shfl_down(pr[hd], off, 32);
    }
  }
  if (lane == 0) {
#pragma unroll
    for (int hd = 0; hd < HEADS; ++hd) {
      el[wave * HEADS + hd] = pl[hd];
      er[wave * HEADS + hd] = pr[hd];
    }
  }
}

// ------------------------------------------------------------- kernel 3 -----
// denom[d,h] += exp(leaky(el[s,h]+er[d,h]))   (max-shift dropped: invariant)
__global__ void k_denom(const int* __restrict__ src, const int* __restrict__ dst,
                        const float* __restrict__ el, const float* __restrict__ er,
                        float* __restrict__ denom, int E) {
  int t = blockIdx.x * blockDim.x + threadIdx.x;
  if (t >= E * HEADS) return;
  int e = t >> 2, hd = t & 3;
  int s = src[e], d = dst[e];
  float x = el[s * HEADS + hd] + er[d * HEADS + hd];
  x = x > 0.0f ? x : NEG_SLOPE * x;
  unsafeAtomicAdd(&denom[d * HEADS + hd], __expf(x));
}

// ------------------------------------------------------------- kernel 4 -----
// out[d,:] += a(e,hd) * h[s,:]   -- one wave per edge, 8 contiguous cols/lane
__global__ void k_aggregate(const int* __restrict__ src, const int* __restrict__ dst,
                            const float* __restrict__ el, const float* __restrict__ er,
                            const float* __restrict__ denom,
                            const float* __restrict__ hmat,
                            float* __restrict__ out, int E) {
  int t = blockIdx.x * blockDim.x + threadIdx.x;
  int e = t >> 5;
  if (e >= E) return;
  int lane = t & 31;
  int hd = lane >> 3;          // 8 lanes per head
  int j0 = lane << 3;          // 8 contiguous columns, all inside head hd
  int s = src[e], d = dst[e];
  float x = el[s * HEADS + hd] + er[d * HEADS + hd];
  x = x > 0.0f ? x : NEG_SLOPE * x;
  float a = __expf(x) / denom[d * HEADS + hd];
  const float4* hp = (const float4*)(hmat + (size_t)s * OUTF + j0);
  float4 v0 = hp[0], v1 = hp[1];
  float* o = out + (size_t)d * OUTF + j0;
  unsafeAtomicAdd(o + 0, v0.x * a);
  unsafeAtomicAdd(o + 1, v0.y * a);
  unsafeAtomicAdd(o + 2, v0.z * a);
  unsafeAtomicAdd(o + 3, v0.w * a);
  unsafeAtomicAdd(o + 4, v1.x * a);
  unsafeAtomicAdd(o + 5, v1.y * a);
  unsafeAtomicAdd(o + 6, v1.z * a);
  unsafeAtomicAdd(o + 7, v1.w * a);
}

// ----------------------------------------------------------------- launch ---
extern "C" void kernel_launch(void* const* d_in, const int* in_sizes, int n_in,
                              void* d_out, int out_size, void* d_ws, size_t ws_size,
                              hipStream_t stream) {
  const float* feat   = (const float*)d_in[0];
  const float* W      = (const float*)d_in[1];
  const float* attn_l = (const float*)d_in[2];
  const float* attn_r = (const float*)d_in[3];
  const float* bias   = (const float*)d_in[4];
  const int*   src    = (const int*)d_in[5];
  const int*   dst    = (const int*)d_in[6];
  float*       out    = (float*)d_out;

  const int n = in_sizes[0] / INFEATS;   // 50000
  const int E = in_sizes[5];             // 800000

  char* ws = (char*)d_ws;
  size_t off = 0;
  float* hmat  = (float*)(ws + off);  off += (size_t)n * OUTF * sizeof(float);
  float* el    = (float*)(ws + off);  off += (size_t)n * HEADS * sizeof(float);
  float* er    = (float*)(ws + off);  off += (size_t)n * HEADS * sizeof(float);
  float* denom = (float*)(ws + off);  off += (size_t)n * HEADS * sizeof(float);
  unsigned short* featb = (unsigned short*)(ws + off);
  off += (size_t)n * INFEATS * sizeof(unsigned short);
  unsigned short* Wb = (unsigned short*)(ws + off);
  off += (size_t)OUTF * INFEATS * sizeof(unsigned short);

  const int B = 256;

  // one-time bf16 conversions (feat: n*256, W: 256*256; both multiples of 8)
  int cf = n * INFEATS / 8;
  k_cvt_bf16<<<(cf + B - 1) / B, B, 0, stream>>>(feat, featb, n * INFEATS);
  int cw = OUTF * INFEATS / 8;
  k_cvt_bf16<<<(cw + B - 1) / B, B, 0, stream>>>(W, Wb, OUTF * INFEATS);

  int ti = n * OUTF;
  k_init<<<(ti + B - 1) / B, B, 0, stream>>>(out, denom, bias, n);

  // GEMM: 4 column strips x ceil(tiles_m/8) row-tile groups, 8 waves/block
  int tiles_m = n >> 4;                        // 3125
  int grid_g = ((tiles_m + 7) / 8) * 4;        // 1564 blocks
  k_gemm_wmma<<<grid_g, B, 0, stream>>>(featb, Wb, hmat, n);

  int ts = n * 32;
  k_scores<<<(ts + B - 1) / B, B, 0, stream>>>(hmat, attn_l, attn_r, el, er, n);

  int td = E * HEADS;
  k_denom<<<(td + B - 1) / B, B, 0, stream>>>(src, dst, el, er, denom, E);

  long long ta = (long long)E * 32;      // 25.6M threads
  k_aggregate<<<(int)((ta + B - 1) / B), B, 0, stream>>>(src, dst, el, er, denom,
                                                         hmat, out, E);
}
